// MultiEntityTransformerEmbModel_44074954392125
// MI455X (gfx1250) — matrix operand order repaired and checked by reading, not compiled, not run on hardware
//
#include <hip/hip_runtime.h>
#include <hip/hip_bf16.h>

// ---------------------------------------------------------------------------
// MultiEntityTransformerEmbModel fused kernel for MI455X (gfx1250, wave32).
//
// kernel 1: pack K/V weights (384x1024 f32) -> bf16 WMMA A-fragments.
// kernel 2: compute Q (static + dynamic GEMV) -> bf16 WMMA A-fragments.
// kernel 3: one workgroup per batch element; flash-style loop over 7
//           key-chunks of 32; per chunk: WMMA bf16 K/V projection with
//           ASYNCcnt-tracked global->LDS double-buffered staging of x,
//           WMMA bf16 scores, f32 online softmax + P@V accumulation.
// ---------------------------------------------------------------------------

typedef __attribute__((ext_vector_type(16))) __bf16 v16bf;
typedef __attribute__((ext_vector_type(8)))  float  v8f;
typedef __attribute__((ext_vector_type(4)))  int    v4i;

#define ASG __attribute__((address_space(1)))
#define ASL __attribute__((address_space(3)))

#if defined(__gfx1250__) && \
    __has_builtin(__builtin_amdgcn_global_load_async_to_lds_b128) && \
    __has_builtin(__builtin_amdgcn_s_wait_asynccnt)
#define USE_ASYNC_COPY 1
#else
#define USE_ASYNC_COPY 0
#endif

#define BN      256
#define CDIM    1024
#define HW      196
#define DM      384
#define NTT     7            // 5 static + 2 dynamic query tokens
#define NKCHUNK 7            // 7 * 32 = 224 >= 196 keys (tail masked)
#define KTILES  12           // 384 / 32 k-steps for the scores GEMM
#define WPACK_ELEMS (24 * 32 * 32 * 16)   // dtiles*ctiles*lanes*elems
#define QPACK_ELEMS (BN * KTILES * 32 * 16)
#define SCALE   0.05103103630798287f      // 1/sqrt(384)

__device__ __forceinline__ unsigned short f32_to_bf16_bits(float f) {
  unsigned int u = __float_as_uint(f);
  u += 0x7FFFu + ((u >> 16) & 1u);        // round-to-nearest-even
  return (unsigned short)(u >> 16);
}
__device__ __forceinline__ float bf16_bits_to_f32(unsigned short h) {
  return __uint_as_float(((unsigned int)h) << 16);
}

#if USE_ASYNC_COPY
// one 16-byte ASYNCcnt-tracked copy: global -> LDS
// builtin signature (from clang diagnostic): (v4i AS1*, v4i AS3*, Imm, Imm)
__device__ __forceinline__ void async_copy_b128(const float* gsrc, float* ldst) {
  float* g = const_cast<float*>(gsrc);
  __builtin_amdgcn_global_load_async_to_lds_b128(
      (ASG v4i*)g, (ASL v4i*)ldst, 0, 0);
}
#endif

// 16-bit A-matrix 16x32 lane layout (contiguous k order, matches B):
// lanes 0-15: M=0..15, elems 0..15 -> K=0..15
// lanes 16-31: same M,  elems 0..15 -> K=16..31
__device__ __forceinline__ int afrag_k(int lane, int e) {
  const int grp = lane >> 4;
  return grp * 16 + e;
}

// ---------------------------------------------------------------------------
// Kernel 1: pack K2d_w / V2d_w into bf16 A fragments.
// Layout: [dtile(24)][ctile(32)][lane(32)][elem(16)] bf16 (32 B/lane contig).
// ---------------------------------------------------------------------------
__global__ __launch_bounds__(256) void pack_weights(
    const float* __restrict__ kw, const float* __restrict__ vw,
    unsigned short* __restrict__ kp, unsigned short* __restrict__ vp) {
  const int idx = blockIdx.x * 256 + threadIdx.x;
  if (idx >= WPACK_ELEMS) return;
  const int e    = idx & 15;
  const int lane = (idx >> 4) & 31;
  const int tile = idx >> 9;
  const int ct   = tile & 31;
  const int dt   = tile >> 5;
  const int d = dt * 16 + (lane & 15);
  const int c = ct * 32 + afrag_k(lane, e);
  kp[idx] = f32_to_bf16_bits(kw[d * CDIM + c]);
  vp[idx] = f32_to_bf16_bits(vw[d * CDIM + c]);
}

// ---------------------------------------------------------------------------
// Kernel 2: build Q (static + dynamic) and pack into bf16 A fragments.
// Layout: [b(256)][ktile(12)][lane(32)][elem(16)] bf16. Rows m>=7 are zero.
// ---------------------------------------------------------------------------
__global__ __launch_bounds__(256) void pack_queries(
    const float* __restrict__ dyn_in, const float* __restrict__ Qs,
    const float* __restrict__ Qsb, const float* __restrict__ dynW,
    const float* __restrict__ dynB, unsigned short* __restrict__ qp) {
  const int idx = blockIdx.x * 256 + threadIdx.x;
  if (idx >= QPACK_ELEMS) return;
  const int e    = idx & 15;
  const int lane = (idx >> 4) & 31;
  const int kt   = (idx >> 9) % KTILES;
  const int b    = (idx >> 9) / KTILES;
  const int m    = lane & 15;               // query-token row (pad to 16)
  const int d    = kt * 32 + afrag_k(lane, e);
  float v = 0.f;
  if (m < 5) {
    v = Qs[m * DM + d] + Qsb[d];
  } else if (m < 7) {
    const int row = (m - 5) * DM + d;
    float s = dynB[row];
    const float* wr = dynW + (size_t)row * CDIM;
    const float* xb = dyn_in + (size_t)b * CDIM;
    for (int c = 0; c < CDIM; ++c) s += xb[c] * wr[c];
    v = s;
  }
  qp[idx] = f32_to_bf16_bits(v);
}

// ---------------------------------------------------------------------------
// Kernel 3: fused K/V projection + attention. grid = 256 (one WG per batch),
// block = 256 (8 wave32 waves).
//   waves 0-3: K projection (6 d-tiles each), waves 4-7: V projection.
// ---------------------------------------------------------------------------
__global__ __launch_bounds__(256) void fused_kv_attn(
    const float* __restrict__ x,
    const float* __restrict__ kbias, const float* __restrict__ vbias,
    const unsigned short* __restrict__ kpack,
    const unsigned short* __restrict__ vpack,
    const unsigned short* __restrict__ qpack,
    float* __restrict__ out) {
#if USE_ASYNC_COPY
  __shared__ alignas(16) float sRaw[2][32][32];           // raw x tiles (dbuf)
#endif
  __shared__ alignas(32) unsigned short sBX[2][32][16];   // x B-fragments
  __shared__ alignas(32) unsigned short sKb[32][DM];      // K chunk [key][d]
  __shared__ alignas(32) unsigned short sVb[32][DM];      // V chunk [key][d]
  __shared__ float sS[16][32];
  __shared__ float sP[NTT][32];
  __shared__ float sOut[NTT][DM];
  __shared__ float sM[NTT], sL[NTT], sAlpha[NTT];

  const int b    = blockIdx.x;
  const int tid  = threadIdx.x;
  const int wave = tid >> 5;
  const int lane = tid & 31;
  const int lgrp = lane >> 4;
  const int l15  = lane & 15;

  for (int i = tid; i < NTT * DM; i += 256) sOut[i / DM][i % DM] = 0.f;
  if (tid < NTT) { sM[tid] = -3.0e38f; sL[tid] = 0.f; }
  __syncthreads();

  const int mat   = wave >> 2;            // 0 = K, 1 = V
  const int dbase = (wave & 3) * 6;       // 6 d-tiles per wave
  const unsigned short* wp  = mat ? vpack : kpack;
  const float* bias         = mat ? vbias : kbias;
  unsigned short (*dstKV)[DM] = mat ? sVb : sKb;

  const float* xb = x + (size_t)b * CDIM * HW;

  // staging geometry: thread t copies/converts row c_local, cols n4..n4+3
  const int c_local = tid >> 3;           // 0..31
  const int n4      = (tid & 7) * 4;      // 0..28 (16 B granules, aligned)

  for (int chunk = 0; chunk < NKCHUNK; ++chunk) {
    const int n0 = chunk * 32;

    // ---- phase A: project 32 keys of K and V over C=1024 ----
    v8f acc[12] = {};

#if USE_ASYNC_COPY
    // preload tile ct=0 into raw buffer 0 (ASYNCcnt-tracked)
    if (n0 + n4 < HW) {
      async_copy_b128(xb + (size_t)c_local * HW + (n0 + n4),
                      &sRaw[0][c_local][n4]);
    }
#endif

    for (int ct = 0; ct < 32; ++ct) {
      const int buf = ct & 1;
#if USE_ASYNC_COPY
      // prefetch next tile into the other buffer, wait for current tile
      if (ct + 1 < 32) {
        if (n0 + n4 < HW) {
          async_copy_b128(
              xb + (size_t)((ct + 1) * 32 + c_local) * HW + (n0 + n4),
              &sRaw[buf ^ 1][c_local][n4]);
        }
        __builtin_amdgcn_s_wait_asynccnt(1);
      } else {
        __builtin_amdgcn_s_wait_asynccnt(0);
      }
      __syncthreads();  // raw tile visible to all waves
      {  // convert raw f32 tile -> bf16 B fragments in LDS
        const int grp = c_local >> 4;
        const int e   = c_local & 15;
        const float4 rv =
            *reinterpret_cast<const float4*>(&sRaw[buf][c_local][n4]);
        const float rvv[4] = {rv.x, rv.y, rv.z, rv.w};
        #pragma unroll
        for (int i = 0; i < 4; ++i) {
          const int nl = n4 + i;
          const float v = (n0 + nl < HW) ? rvv[i] : 0.f;
          sBX[nl >> 4][(grp << 4) | (nl & 15)][e] = f32_to_bf16_bits(v);
        }
      }
#else
      {  // fallback: direct global->VGPR->LDS staging
        const int c0  = ct * 32;
        const int grp = c_local >> 4;
        const int e   = c_local & 15;
        const float* xr = xb + (size_t)(c0 + c_local) * HW;
        #pragma unroll
        for (int i = 0; i < 4; ++i) {
          const int nl = n4 + i;
          const int n  = n0 + nl;
          const float v = (n < HW) ? xr[n] : 0.f;
          sBX[nl >> 4][(grp << 4) | (nl & 15)][e] = f32_to_bf16_bits(v);
        }
      }
#endif
      __syncthreads();  // sBX fragments ready

      const v16bf b0 = *reinterpret_cast<const v16bf*>(&sBX[0][lane][0]);
      const v16bf b1 = *reinterpret_cast<const v16bf*>(&sBX[1][lane][0]);
      if (ct + 1 < 32) {  // keep next weight fragments warm (global_prefetch)
        __builtin_prefetch(
            (const void*)(wp + ((((dbase * 32 + (ct + 1)) * 32) + lane) << 4)),
            0, 0);
      }
      #pragma unroll
      for (int j = 0; j < 6; ++j) {
        const int dt = dbase + j;
        const v16bf a = *reinterpret_cast<const v16bf*>(
            wp + ((((dt * 32 + ct) * 32) + lane) << 4));
        acc[2 * j] = __builtin_amdgcn_wmma_f32_16x16x32_bf16(
            false, a, false, b0, (short)0, acc[2 * j], false, false);
        acc[2 * j + 1] = __builtin_amdgcn_wmma_f32_16x16x32_bf16(
            false, a, false, b1, (short)0, acc[2 * j + 1], false, false);
      }
      __syncthreads();  // before next iteration overwrites sBX
    }

    // spill K/V chunk (+bias) to LDS as bf16 [key][d]
    // C/D layout: VGPR r -> M = r + 8*(lane>=16), N = lane&15
    #pragma unroll
    for (int j = 0; j < 6; ++j) {
      #pragma unroll
      for (int nt = 0; nt < 2; ++nt) {
        #pragma unroll
        for (int r = 0; r < 8; ++r) {
          const int d   = (dbase + j) * 16 + (lgrp << 3) + r;
          const int key = nt * 16 + l15;
          dstKV[key][d] = f32_to_bf16_bits(acc[2 * j + nt][r] + bias[d]);
        }
      }
    }
    __syncthreads();

    // ---- phase B: scores = Q @ K^T (waves 0,1; one 16x16 tile each) ----
    if (wave < 2) {
      const int nt = wave;
      const int key = nt * 16 + l15;
      v8f sc = {};
      #pragma unroll
      for (int kt = 0; kt < KTILES; ++kt) {
        const v16bf aq = *reinterpret_cast<const v16bf*>(
            qpack + (((size_t)b * KTILES + kt) * 32 + lane) * 16);
        // 16 consecutive d values for fixed key: one 32 B LDS vector load
        const v16bf bk = *reinterpret_cast<const v16bf*>(
            &sKb[key][kt * 32 + (lgrp << 4)]);
        sc = __builtin_amdgcn_wmma_f32_16x16x32_bf16(
            false, aq, false, bk, (short)0, sc, false, false);
      }
      #pragma unroll
      for (int r = 0; r < 8; ++r)
        sS[(lgrp << 3) + r][nt * 16 + l15] = sc[r];
    }
    __syncthreads();

    // ---- phase C: online softmax for the 7 real query rows ----
    if (tid < NTT) {
      float mx = -3.0e38f;
      for (int j = 0; j < 32; ++j) {
        const float s = (n0 + j < HW) ? sS[tid][j] * SCALE : -3.0e38f;
        sS[tid][j] = s;
        mx = fmaxf(mx, s);
      }
      const float m_old = sM[tid];
      const float m_new = fmaxf(m_old, mx);
      const float alpha = __expf(m_old - m_new);
      float psum = 0.f;
      for (int j = 0; j < 32; ++j) {
        const float s = sS[tid][j];
        const float p = (s > -1.0e37f) ? __expf(s - m_new) : 0.f;
        sP[tid][j] = p;
        psum += p;
      }
      sM[tid] = m_new;
      sL[tid] = sL[tid] * alpha + psum;
      sAlpha[tid] = alpha;
    }
    __syncthreads();

    // ---- phase D: out = out*alpha + P @ V_chunk (f32, tiny) ----
    for (int idx = tid; idx < NTT * DM; idx += 256) {
      const int t = idx / DM;
      const int d = idx % DM;
      float s = 0.f;
      #pragma unroll 4
      for (int j = 0; j < 32; ++j) s += sP[t][j] * bf16_bits_to_f32(sVb[j][d]);
      sOut[t][d] = sOut[t][d] * sAlpha[t] + s;
    }
    __syncthreads();
  }

  // final: normalize and write transposed out[b][d][t]
  for (int idx = tid; idx < NTT * DM; idx += 256) {
    const int t = idx / DM;
    const int d = idx % DM;
    out[((size_t)b * DM + d) * NTT + t] = sOut[t][d] / sL[t];
  }
}

// ---------------------------------------------------------------------------
extern "C" void kernel_launch(void* const* d_in, const int* in_sizes, int n_in,
                              void* d_out, int out_size, void* d_ws, size_t ws_size,
                              hipStream_t stream) {
  const float* x      = (const float*)d_in[0];
  const float* dyn_in = (const float*)d_in[1];
  const float* K2d_w  = (const float*)d_in[2];
  const float* K2d_b  = (const float*)d_in[3];
  const float* V2d_w  = (const float*)d_in[4];
  const float* V2d_b  = (const float*)d_in[5];
  const float* Q_s    = (const float*)d_in[6];
  const float* Q_s_b  = (const float*)d_in[7];
  const float* dynQ_w = (const float*)d_in[8];
  const float* dynQ_b = (const float*)d_in[9];
  float* out = (float*)d_out;

  unsigned short* kpack = (unsigned short*)d_ws;
  unsigned short* vpack = kpack + WPACK_ELEMS;
  unsigned short* qpack = vpack + WPACK_ELEMS;
  // ws bytes needed: 2*WPACK + QPACK = 2*786432 + 3145728 = 4,718,592 B

  pack_weights<<<(WPACK_ELEMS + 255) / 256, 256, 0, stream>>>(
      K2d_w, V2d_w, kpack, vpack);
  pack_queries<<<(QPACK_ELEMS + 255) / 256, 256, 0, stream>>>(
      dyn_in, Q_s, Q_s_b, dynQ_w, dynQ_b, qpack);
  fused_kv_attn<<<BN, 256, 0, stream>>>(
      x, K2d_b, V2d_b, kpack, vpack, qpack, out);
}
